// ClusteredLinear_266287972671
// MI455X (gfx1250) — matrix operand order repaired and checked by reading, not compiled, not run on hardware
//
#include <hip/hip_runtime.h>
#include <hip/hip_bf16.h>
#include <stdint.h>

// ---------------------------------------------------------------------------
// ClusteredLinear on MI455X (gfx1250): per-channel GEMM 64x720 @ 720x336
//   out[b,c,p] = sum_s x[b,c,s] * W[clu(c),p,s] + bias[clu(c),p]
//
// Pass 1: convert W (8x336x720 f32, 7.7MB) -> bf16 in d_ws (3.87MB, L2-resident)
// Pass 2: WMMA GEMM. Grid (862 channels). Block 224 = 7 waves.
//         Each block: full M=64 batch x N=336; each wave: 64(M) x 48(N).
//         K=720 zero-padded to 736 = 23*32. Per k-step per wave:
//         8 ds_load_b128 (4 A frags) + 6 global_load_b128 (3 bf16 B frags)
//         + 12 v_wmma_f32_16x16x32_bf16  (0.5 global loads per WMMA).
// ---------------------------------------------------------------------------

typedef __attribute__((ext_vector_type(16))) __bf16 v16bf;
typedef __attribute__((ext_vector_type(8)))  float  v8f;

constexpr int SEQ  = 720;   // K
constexpr int PRED = 336;   // N per channel
constexpr int CH   = 862;
constexpr int KPAD = 736;   // 23 * 32
constexpr int LDS_ROW_DW = 372;       // 744 bf16 per row; 372%64=52 -> bank-friendly
constexpr int W_ROW_DW   = SEQ / 2;   // 360 dwords per bf16 W row
constexpr int LDS_BYTES  = 64 * LDS_ROW_DW * 4;  // 93 KB (dynamic LDS)

union Frag {
    v16bf    v;
    uint32_t u[8];
    uint4    q[2];
};

__device__ __forceinline__ uint32_t pk_bf16(float a, float b) {
    uint32_t ua = __float_as_uint(a);
    uint32_t ub = __float_as_uint(b);
    ua += 0x7FFFu + ((ua >> 16) & 1u);   // round-to-nearest-even truncation
    ub += 0x7FFFu + ((ub >> 16) & 1u);
    return (ua >> 16) | (ub & 0xFFFF0000u);
}

__device__ __forceinline__ v8f wmma_bf16(const Frag& a, const Frag& b, v8f c) {
    return __builtin_amdgcn_wmma_f32_16x16x32_bf16(
        false, a.v, false, b.v, (short)0, c, false, false);
}

// Fallback B load: 16 consecutive f32 -> 8 packed bf16 dwords.
__device__ __forceinline__ void load_b_f32(Frag& f, const float* __restrict__ w) {
    const float4* w4 = reinterpret_cast<const float4*>(w);
    float4 f0 = w4[0], f1 = w4[1], f2 = w4[2], f3 = w4[3];
    f.u[0] = pk_bf16(f0.x, f0.y); f.u[1] = pk_bf16(f0.z, f0.w);
    f.u[2] = pk_bf16(f1.x, f1.y); f.u[3] = pk_bf16(f1.z, f1.w);
    f.u[4] = pk_bf16(f2.x, f2.y); f.u[5] = pk_bf16(f2.z, f2.w);
    f.u[6] = pk_bf16(f3.x, f3.y); f.u[7] = pk_bf16(f3.z, f3.w);
}

// Fast B load: 16 consecutive bf16 (32B) -> two b128 loads, no conversion.
__device__ __forceinline__ void load_b_bf16(Frag& f, const uint32_t* __restrict__ w) {
    f.q[0] = *reinterpret_cast<const uint4*>(w);
    f.q[1] = *reinterpret_cast<const uint4*>(w + 4);
}

// ---- Pass 1: W f32 -> bf16 (pairs packed into dwords) -----------------------
__global__ __launch_bounds__(256)
void convert_w_kernel(const float* __restrict__ W, uint32_t* __restrict__ Wbf, int npairs) {
    int i = blockIdx.x * 256 + threadIdx.x;
    if (i < npairs) {
        float2 f = reinterpret_cast<const float2*>(W)[i];
        Wbf[i] = pk_bf16(f.x, f.y);
    }
}

// ---- Pass 2: WMMA GEMM ------------------------------------------------------
template <bool WBF16>
__global__ __launch_bounds__(224)
void clustered_linear_wmma(const float*    __restrict__ x,
                           const int*      __restrict__ clusters,
                           const float*    __restrict__ W,
                           const uint32_t* __restrict__ Wbf,
                           const float*    __restrict__ bias,
                           float*          __restrict__ out) {
    const int c   = blockIdx.x;          // channel
    const int clu = clusters[c];

    extern __shared__ uint32_t ldsA[];   // 64 rows x 744 bf16 = 93 KB

    const int tid = threadIdx.x;

    // Stage A (64 batches x 720 K) into LDS as bf16, zero-padded to 736.
    for (int i = tid; i < 64 * (KPAD / 2); i += 224) {
        int row  = i / (KPAD / 2);
        int dcol = i % (KPAD / 2);
        int s    = dcol * 2;
        float a0 = 0.0f, a1 = 0.0f;
        if (s < SEQ) {
            const float* xr = x + ((size_t)row * CH + c) * SEQ + s;
            a0 = xr[0];
            a1 = xr[1];
        }
        ldsA[row * LDS_ROW_DW + dcol] = pk_bf16(a0, a1);
    }
    __syncthreads();

    const int lane  = tid & 31;
    const int wave  = tid >> 5;          // 0..6
    const int lrow  = lane & 15;
    const int khalf = lane >> 4;

    // A fragments: lane<16 holds K {0..15}, lane>=16 {16..31}.
    const uint32_t* aBase[4];
    #pragma unroll
    for (int mt = 0; mt < 4; ++mt)
        aBase[mt] = ldsA + (lrow + 16 * mt) * LDS_ROW_DW + khalf * 8;

    // B: lane holds column p0+16*nt; lanes 0-15 K=k0..k0+15, 16-31 K=k0+16..k0+31.
    const int p0 = wave * 48 + lrow;
    const float*    wf[3];
    const uint32_t* wb[3];
    #pragma unroll
    for (int nt = 0; nt < 3; ++nt) {
        wf[nt] = W   + ((size_t)clu * PRED + p0 + 16 * nt) * SEQ      + khalf * 16;
        wb[nt] = Wbf + ((size_t)clu * PRED + p0 + 16 * nt) * W_ROW_DW + khalf * 8;
    }

    v8f acc[4][3];
    #pragma unroll
    for (int mt = 0; mt < 4; ++mt)
        #pragma unroll
        for (int nt = 0; nt < 3; ++nt)
            acc[mt][nt] = (v8f){};

    Frag a[4], b;

    // Main K loop: 22 unguarded steps of 32.
    for (int k0 = 0; k0 < 704; k0 += 32) {
        #pragma unroll
        for (int mt = 0; mt < 4; ++mt) {
            a[mt].q[0] = *reinterpret_cast<const uint4*>(aBase[mt] + (k0 >> 1));
            a[mt].q[1] = *reinterpret_cast<const uint4*>(aBase[mt] + (k0 >> 1) + 4);
        }
        #pragma unroll
        for (int nt = 0; nt < 3; ++nt) {
            if (WBF16) load_b_bf16(b, wb[nt] + (k0 >> 1));
            else       load_b_f32 (b, wf[nt] + k0);
            #pragma unroll
            for (int mt = 0; mt < 4; ++mt)
                acc[mt][nt] = wmma_bf16(a[mt], b, acc[mt][nt]);
        }
    }

    // Tail k0=704: A reads zero-padded LDS; B valid only for lanes<16 (s=704..719).
    {
        const int k0 = 704;
        #pragma unroll
        for (int mt = 0; mt < 4; ++mt) {
            a[mt].q[0] = *reinterpret_cast<const uint4*>(aBase[mt] + (k0 >> 1));
            a[mt].q[1] = *reinterpret_cast<const uint4*>(aBase[mt] + (k0 >> 1) + 4);
        }
        #pragma unroll
        for (int nt = 0; nt < 3; ++nt) {
            Frag bz;
            if (khalf == 0) {
                if (WBF16) load_b_bf16(bz, wb[nt] + (k0 >> 1));
                else       load_b_f32 (bz, wf[nt] + k0);
            } else {
                #pragma unroll
                for (int j = 0; j < 8; ++j) bz.u[j] = 0u;
            }
            #pragma unroll
            for (int mt = 0; mt < 4; ++mt)
                acc[mt][nt] = wmma_bf16(a[mt], bz, acc[mt][nt]);
        }
    }

    // Bias + store. D layout: VGPR j holds M = khalf*8 + j, N = lrow.
    float bv[3];
    #pragma unroll
    for (int nt = 0; nt < 3; ++nt)
        bv[nt] = bias[clu * PRED + p0 + 16 * nt];

    #pragma unroll
    for (int mt = 0; mt < 4; ++mt) {
        #pragma unroll
        for (int j = 0; j < 8; ++j) {
            int m = 16 * mt + khalf * 8 + j;
            float* o = out + ((size_t)m * CH + c) * PRED + p0;
            o[ 0] = acc[mt][0][j] + bv[0];
            o[16] = acc[mt][1][j] + bv[1];
            o[32] = acc[mt][2][j] + bv[2];
        }
    }
}

extern "C" void kernel_launch(void* const* d_in, const int* in_sizes, int n_in,
                              void* d_out, int out_size, void* d_ws, size_t ws_size,
                              hipStream_t stream) {
    const float* x        = (const float*)d_in[0];
    const int*   clusters = (const int*)  d_in[1];
    const float* W        = (const float*)d_in[2];
    const float* b        = (const float*)d_in[3];
    float*       out      = (float*)d_out;

    const int    npairs = 8 * PRED * SEQ / 2;          // 967,680 packed dwords
    const size_t needW  = (size_t)npairs * 4;          // 3.87 MB

    if (ws_size >= needW) {
        uint32_t* Wbf = (uint32_t*)d_ws;
        convert_w_kernel<<<(npairs + 255) / 256, 256, 0, stream>>>(W, Wbf, npairs);
        clustered_linear_wmma<true><<<CH, 224, LDS_BYTES, stream>>>(x, clusters, W, Wbf, b, out);
    } else {
        clustered_linear_wmma<false><<<CH, 224, LDS_BYTES, stream>>>(x, clusters, W, nullptr, b, out);
    }
}
